// WMSA_16484084483155
// MI455X (gfx1250) — matrix operand order repaired and checked by reading, not compiled, not run on hardware
//
#include <hip/hip_runtime.h>

// ---------------------------------------------------------------------------
// Fused shifted-window MSA for MI455X (gfx1250, wave32, WMMA bf16).
// One workgroup (8 waves) per 8x8 window; all intermediates in LDS.
// HBM traffic ~= read x (256MB) + write out (256MB); weights stream from L2.
// ---------------------------------------------------------------------------

typedef __attribute__((ext_vector_type(16))) __bf16 v16bf;
typedef __attribute__((ext_vector_type(8)))  float  v8f;

union FragB16 {
    v16bf v;
    unsigned int u[8];
    uint4 q[2];
};

__device__ __forceinline__ unsigned short f2bf(float f) {
    unsigned int u = __float_as_uint(f);
    unsigned int r = u + 0x7FFFu + ((u >> 16) & 1u);   // round-to-nearest-even
    return (unsigned short)(r >> 16);
}

__device__ __forceinline__ v8f zero8() {
    v8f z = {0.f, 0.f, 0.f, 0.f, 0.f, 0.f, 0.f, 0.f};
    return z;
}

__device__ __forceinline__ v8f mma_bf16(FragB16 a, FragB16 b, v8f c) {
    return __builtin_amdgcn_wmma_f32_16x16x32_bf16(
        /*neg_a=*/false, a.v, /*neg_b=*/false, b.v,
        /*c_mod=*/(short)0, c, /*reuse_a=*/false, /*reuse_b=*/false);
}

// ---------------- kernel 0: fp32 weights -> bf16 (row-major kept) ----------
__global__ void wmsa_cvt_weights(const float* __restrict__ wqkv,
                                 const float* __restrict__ wout,
                                 unsigned short* __restrict__ wqkv_bf,
                                 unsigned short* __restrict__ wout_bf) {
    int i = blockIdx.x * 256 + threadIdx.x;
    if (i < 768 * 256) wqkv_bf[i] = f2bf(wqkv[i]);
    if (i < 256 * 256) wout_bf[i] = f2bf(wout[i]);
}

// ---------------- fused window-attention kernel ----------------------------
// LDS layout (bytes):
//   [0      , 32768 )  x tile   bf16 [64][256]
//   [32768  , 131072)  qkv      bf16 [64][768]   (q scaled; cols 0..255 q, 256..511 k, 512..767 v)
//   [131072 , 196608)  P        bf16 [8 heads][64][64]
//   [196608 , 229376)  attn out bf16 [64][256]
//   [229376 , 236576)  rel_pos  f32  [8][15*15]
#define SMEM_BYTES 236800

__global__ void __launch_bounds__(256)
wmsa_fused(const float* __restrict__ x,
           const float* __restrict__ b_qkv,
           const float* __restrict__ rel_pos,
           const float* __restrict__ b_out,
           const unsigned short* __restrict__ wqkv_bf,
           const unsigned short* __restrict__ wout_bf,
           float* __restrict__ out) {
    extern __shared__ char smem[];
    unsigned short* lds_x   = (unsigned short*)(smem);
    unsigned short* lds_qkv = (unsigned short*)(smem + 32768);
    unsigned short* lds_p   = (unsigned short*)(smem + 131072);
    unsigned short* lds_ao  = (unsigned short*)(smem + 196608);
    float*          lds_rel = (float*)        (smem + 229376);

    const int tid  = threadIdx.x;
    const int wave = tid >> 5;
    const int lane = tid & 31;
    const int hsel = lane >> 4;     // which 16-lane half
    const int lx   = lane & 15;

    const int bw = blockIdx.x;      // 0..4095
    const int b  = bw >> 10;
    const int w  = bw & 1023;
    const int wy = w >> 5, wx = w & 31;
    const bool lastrow = (wy == 31), lastcol = (wx == 31);

    // A fragment: 16x32 bf16 tile, row-major source with `stride` elements.
    auto loadA = [&](const unsigned short* base, int stride, int m0, int k0) {
        FragB16 f;
        const unsigned short* r = base + (m0 + lx) * stride + k0 + hsel * 16;
        f.q[0] = *(const uint4*)(r);        // K = koff .. koff+7
        f.q[1] = *(const uint4*)(r + 8);    // K = koff+8 .. koff+15
        return f;
    };
    // B fragment: 32x16, element (K,N) = base[(n0+N)*stride + k0+K] (contiguous K).
    auto loadBrow = [&](const unsigned short* base, int stride, int n0, int k0) {
        FragB16 f;
        const unsigned short* r = base + (n0 + lx) * stride + k0 + hsel * 16;
        f.q[0] = *(const uint4*)(r);        // K = koff .. koff+7
        f.q[1] = *(const uint4*)(r + 8);    // K = koff+8 .. koff+15
        return f;
    };
    // B fragment: 32x16, element (K,N) = base[(k0+K)*stride + n0+N] (strided K).
    auto loadBcol = [&](const unsigned short* base, int stride, int n0, int k0) {
        FragB16 f;
        const unsigned short* c = base + (k0 + hsel * 16) * stride + n0 + lx;
        #pragma unroll
        for (int j = 0; j < 8; ++j)
            f.u[j] = (unsigned int)c[(2 * j) * stride] |
                     ((unsigned int)c[(2 * j + 1) * stride] << 16);
        return f;
    };

    // ---- preload relative-position table ----
    for (int i = tid; i < 8 * 225; i += 256) lds_rel[i] = rel_pos[i];

    // ---- gather rolled x tile, convert to bf16 ----
    {
        int p  = tid >> 2;            // token 0..63
        int c4 = (tid & 3) * 64;      // channel base
        int py = p >> 3, px = p & 7;
        int row = (wy * 8 + py + 4) & 255;
        int col = (wx * 8 + px + 4) & 255;
        const float4* src =
            (const float4*)(x + (((long)(b * 256 + row) * 256 + col) << 8) + c4);
        unsigned int* dst = (unsigned int*)(lds_x + p * 256 + c4);
        #pragma unroll
        for (int i = 0; i < 16; ++i) {
            float4 f = src[i];
            dst[i * 2 + 0] = (unsigned int)f2bf(f.x) | ((unsigned int)f2bf(f.y) << 16);
            dst[i * 2 + 1] = (unsigned int)f2bf(f.z) | ((unsigned int)f2bf(f.w) << 16);
        }
    }
    __syncthreads();

    // ---- stage 1: qkv = x_tile(64x256) @ wqkv^T(256x768) + bias, q *= scale ----
    {
        const int wcol = wave * 96;   // 8 waves x 96 cols = 768
        for (int nt = 0; nt < 6; ++nt) {
            v8f acc[4];
            #pragma unroll
            for (int mt = 0; mt < 4; ++mt) acc[mt] = zero8();
            for (int kc = 0; kc < 8; ++kc) {
                FragB16 bfr = loadBrow(wqkv_bf, 256, wcol + nt * 16, kc * 32);
                #pragma unroll
                for (int mt = 0; mt < 4; ++mt) {
                    FragB16 afr = loadA(lds_x, 256, mt * 16, kc * 32);
                    acc[mt] = mma_bf16(afr, bfr, acc[mt]);
                }
            }
            const int n    = wcol + nt * 16 + lx;
            const float bi = b_qkv[n];
            const float sc = (n < 256) ? 0.17677669529663687f : 1.0f;   // 1/sqrt(32)
            #pragma unroll
            for (int mt = 0; mt < 4; ++mt) {
                #pragma unroll
                for (int r = 0; r < 8; ++r) {
                    int m = mt * 16 + hsel * 8 + r;
                    lds_qkv[m * 768 + n] = f2bf((acc[mt][r] + bi) * sc);
                }
            }
        }
    }
    __syncthreads();

    // ---- stage 2: per-wave head attention ----
    {
        const int h  = wave;
        const int qb = h * 32, kb = 256 + h * 32, vb = 512 + h * 32;

        FragB16 aq[4];
        #pragma unroll
        for (int mt = 0; mt < 4; ++mt) aq[mt] = loadA(lds_qkv, 768, mt * 16, qb);

        v8f s[4][4];
        #pragma unroll
        for (int nt = 0; nt < 4; ++nt) {
            FragB16 bk = loadBrow(lds_qkv, 768, nt * 16, kb);   // k^T : (K,N)=k[N][K]
            #pragma unroll
            for (int mt = 0; mt < 4; ++mt)
                s[mt][nt] = mma_bf16(aq[mt], bk, zero8());
        }

        // relative bias + shift mask
        #pragma unroll
        for (int mt = 0; mt < 4; ++mt) {
            #pragma unroll
            for (int r = 0; r < 8; ++r) {
                int p  = mt * 16 + hsel * 8 + r;
                int pi = p >> 3, pj = p & 7;
                #pragma unroll
                for (int nt = 0; nt < 4; ++nt) {
                    int qt = nt * 16 + lx;
                    int qi = qt >> 3, qj = qt & 7;
                    float v = s[mt][nt][r] +
                              lds_rel[h * 225 + (pi - qi + 7) * 15 + (pj - qj + 7)];
                    bool msk = (lastrow && ((pi >> 2) != (qi >> 2))) ||
                               (lastcol && ((pj >> 2) != (qj >> 2)));
                    s[mt][nt][r] = msk ? -1e30f : v;
                }
            }
        }

        // softmax over N (row lives in one VGPR across a 16-lane half)
        #pragma unroll
        for (int mt = 0; mt < 4; ++mt) {
            #pragma unroll
            for (int r = 0; r < 8; ++r) {
                float mx = fmaxf(fmaxf(s[mt][0][r], s[mt][1][r]),
                                 fmaxf(s[mt][2][r], s[mt][3][r]));
                #pragma unroll
                for (int d = 1; d < 16; d <<= 1) mx = fmaxf(mx, __shfl_xor(mx, d));
                float e[4], sum = 0.f;
                #pragma unroll
                for (int nt = 0; nt < 4; ++nt) {
                    e[nt] = __expf(s[mt][nt][r] - mx);
                    sum += e[nt];
                }
                #pragma unroll
                for (int d = 1; d < 16; d <<= 1) sum += __shfl_xor(sum, d);
                float inv = __builtin_amdgcn_rcpf(sum);
                int p = mt * 16 + hsel * 8 + r;
                unsigned short* prow = lds_p + (h * 64 + p) * 64;
                #pragma unroll
                for (int nt = 0; nt < 4; ++nt)
                    prow[nt * 16 + lx] = f2bf(e[nt] * inv);
            }
        }

        // out_h(64x32) = P(64x64) @ V(64x32)
        v8f o[4][2];
        #pragma unroll
        for (int mt = 0; mt < 4; ++mt)
            #pragma unroll
            for (int n = 0; n < 2; ++n) o[mt][n] = zero8();
        for (int kc = 0; kc < 2; ++kc) {
            FragB16 ap[4];
            #pragma unroll
            for (int mt = 0; mt < 4; ++mt)
                ap[mt] = loadA(lds_p + h * 4096, 64, mt * 16, kc * 32);
            #pragma unroll
            for (int n = 0; n < 2; ++n) {
                FragB16 bv = loadBcol(lds_qkv, 768, vb + n * 16, kc * 32);
                #pragma unroll
                for (int mt = 0; mt < 4; ++mt)
                    o[mt][n] = mma_bf16(ap[mt], bv, o[mt][n]);
            }
        }
        #pragma unroll
        for (int mt = 0; mt < 4; ++mt) {
            #pragma unroll
            for (int n = 0; n < 2; ++n) {
                #pragma unroll
                for (int r = 0; r < 8; ++r) {
                    int m = mt * 16 + hsel * 8 + r;
                    lds_ao[m * 256 + h * 32 + n * 16 + lx] = f2bf(o[mt][n][r]);
                }
            }
        }
    }
    __syncthreads();

    // ---- stage 3: out = ao(64x256) @ wout^T(256x256) + b_out, rolled scatter ----
    {
        const int c0 = wave * 32;     // 8 waves x 32 cols = 256
        v8f o[4][2];
        #pragma unroll
        for (int mt = 0; mt < 4; ++mt)
            #pragma unroll
            for (int n = 0; n < 2; ++n) o[mt][n] = zero8();
        for (int kc = 0; kc < 8; ++kc) {
            FragB16 af[4];
            #pragma unroll
            for (int mt = 0; mt < 4; ++mt)
                af[mt] = loadA(lds_ao, 256, mt * 16, kc * 32);
            #pragma unroll
            for (int n = 0; n < 2; ++n) {
                FragB16 bfr = loadBrow(wout_bf, 256, c0 + n * 16, kc * 32);
                #pragma unroll
                for (int mt = 0; mt < 4; ++mt)
                    o[mt][n] = mma_bf16(af[mt], bfr, o[mt][n]);
            }
        }
        #pragma unroll
        for (int n = 0; n < 2; ++n) {
            const int ch  = c0 + n * 16 + lx;
            const float bi = b_out[ch];
            #pragma unroll
            for (int mt = 0; mt < 4; ++mt) {
                #pragma unroll
                for (int r = 0; r < 8; ++r) {
                    int p  = mt * 16 + hsel * 8 + r;
                    int py = p >> 3, px = p & 7;
                    int row = (wy * 8 + py + 4) & 255;
                    int col = (wx * 8 + px + 4) & 255;
                    out[(((long)(b * 256 + row) * 256 + col) << 8) + ch] =
                        o[mt][n][r] + bi;
                }
            }
        }
    }
}

extern "C" void kernel_launch(void* const* d_in, const int* in_sizes, int n_in,
                              void* d_out, int out_size, void* d_ws, size_t ws_size,
                              hipStream_t stream) {
    const float* x     = (const float*)d_in[0];
    const float* w_qkv = (const float*)d_in[1];
    const float* bqkv  = (const float*)d_in[2];
    const float* rel   = (const float*)d_in[3];
    const float* w_out = (const float*)d_in[4];
    const float* bout  = (const float*)d_in[5];

    unsigned short* wqkv_bf = (unsigned short*)d_ws;                       // 768*256 bf16
    unsigned short* wout_bf = (unsigned short*)((char*)d_ws + 768 * 256 * 2); // 256*256 bf16

    hipLaunchKernelGGL(wmsa_cvt_weights, dim3(768), dim3(256), 0, stream,
                       w_qkv, w_out, wqkv_bf, wout_bf);

    hipLaunchKernelGGL(wmsa_fused, dim3(4096), dim3(256), SMEM_BYTES, stream,
                       x, bqkv, rel, bout, wqkv_bf, wout_bf, (float*)d_out);
}
